// Translator_49065706389863
// MI455X (gfx1250) — compile-verified, matrix operand order checked by reading, not run on hardware
//
#include <hip/hip_runtime.h>
#include <hip/hip_bf16.h>
#include <math.h>

// ---------------- problem constants (from reference) ----------------
#define BATCH   16
#define SRCLEN  512
#define HID     768
#define VOCAB   30522
#define VPAD    30528              // round up to 16
#define NBEAM   5
#define MAXLEN  64
#define MINLEN  10
#define BOSTOK  101
#define EOSTOK  102
#define NEGBIG  (-1e20f)
#define ALPHA_P 0.6f
#define ROWS    (BATCH*NBEAM)      // 80
#define NTILE   (VPAD/16)          // 1908
#define KBLK    (HID/32)           // 24

typedef __attribute__((ext_vector_type(16))) __bf16 v16bf;
typedef __attribute__((ext_vector_type(8)))  __bf16 v8bf;
typedef __attribute__((ext_vector_type(8)))  float  v8f;

__device__ __forceinline__ unsigned short f32_to_bf16(float f) {
    unsigned int u = __float_as_uint(f);
    unsigned int lsb = (u >> 16) & 1u;
    u += 0x7fffu + lsb;            // round-to-nearest-even
    return (unsigned short)(u >> 16);
}

// ---------------- one-time prep kernels ----------------

// src_feats[b][s][h] = embed[src[b,s]][h] * mask
__global__ void k_src_feats(const int* __restrict__ src, const int* __restrict__ mask,
                            const float* __restrict__ embed, float* __restrict__ sf) {
    size_t idx = (size_t)blockIdx.x * 256 + threadIdx.x;
    size_t total = (size_t)BATCH * SRCLEN * HID;
    if (idx >= total) return;
    int h = (int)(idx % HID);
    size_t t = idx / HID;
    int s = (int)(t % SRCLEN);
    int b = (int)(t / SRCLEN);
    int tok = src[b * SRCLEN + s];
    float m = (mask[b * SRCLEN + s] != 0) ? 1.0f : 0.0f;
    sf[idx] = embed[(size_t)tok * HID + h] * m;
}

// Pack W_gen (HID x VOCAB, f32 row-major) into per-lane WMMA-B bf16 fragments:
// Wb[((nt*KBLK + kb)*32 + lane)*16 + e],
// lane&15 = output column within 16-wide N tile, lane>>4 selects K-half,
// e<8 -> K = kb*32 + half*8 + e ; e>=8 -> K = kb*32 + half*8 + 8 + e
__global__ void k_pack_w(const float* __restrict__ W, unsigned short* __restrict__ Wb) {
    size_t idx = (size_t)blockIdx.x * 256 + threadIdx.x;
    size_t total = (size_t)NTILE * KBLK * 32 * 16;
    if (idx >= total) return;
    int e    = (int)(idx & 15);
    size_t t = idx >> 4;
    int lane = (int)(t & 31);
    t >>= 5;
    int kb = (int)(t % KBLK);
    int nt = (int)(t / KBLK);
    int half = lane >> 4;
    int k = kb * 32 + half * 8 + e + ((e >= 8) ? 8 : 0);
    int col = nt * 16 + (lane & 15);
    float v = (col < VOCAB) ? W[(size_t)k * VOCAB + col] : 0.0f;
    Wb[idx] = f32_to_bf16(v);
}

__global__ void k_init(int* __restrict__ aliveA, float* __restrict__ tlp,
                       int* __restrict__ done, float* __restrict__ bestS,
                       int* __restrict__ bestP) {
    int tid = blockIdx.x * 256 + threadIdx.x;
    int total = ROWS * (MAXLEN + 1);
    if (tid < total) {
        int c = tid % (MAXLEN + 1);
        aliveA[tid] = (c == 0) ? BOSTOK : 0;
    }
    if (tid < ROWS)  tlp[tid] = ((tid % NBEAM) == 0) ? 0.0f : -INFINITY;
    if (tid < BATCH) { done[tid] = 0; bestS[tid] = -INFINITY; }
    if (tid < BATCH * MAXLEN) bestP[tid] = 0;
}

// ---------------- per-step kernels ----------------

__global__ void k_gather_x(const int* __restrict__ aliveCur, const float* __restrict__ embed,
                           float* __restrict__ x, int step) {
    int row = blockIdx.x;
    int tok = aliveCur[row * (MAXLEN + 1) + step];
    const float* e = embed + (size_t)tok * HID;
    for (int h = threadIdx.x; h < HID; h += 256)
        x[(size_t)row * HID + h] = e[h];
}

// attention + softmax + context + tanh  ->  dec in bf16 (one block per (b,k) row)
__global__ void k_attn(const float* __restrict__ x, const float* __restrict__ srcFeats,
                       const int* __restrict__ mask, unsigned short* __restrict__ decb) {
    int blk = blockIdx.x;              // 0..79
    int b = blk / NBEAM;
    int tid = threadIdx.x;
    __shared__ float sAtt[SRCLEN];
    __shared__ float red[256];
    const float scale = rsqrtf((float)HID);
    const float* xr = x + (size_t)blk * HID;
    const float* sf = srcFeats + (size_t)b * SRCLEN * HID;

    for (int s = tid; s < SRCLEN; s += 256) {
        const float* sr = sf + (size_t)s * HID;
        float d = 0.0f;
        for (int h = 0; h < HID; ++h) d += xr[h] * sr[h];
        float bias = (mask[b * SRCLEN + s] == 0) ? NEGBIG : 0.0f;
        sAtt[s] = d * scale + bias;
    }
    __syncthreads();
    float m = -INFINITY;
    for (int s = tid; s < SRCLEN; s += 256) m = fmaxf(m, sAtt[s]);
    red[tid] = m; __syncthreads();
    for (int st = 128; st > 0; st >>= 1) { if (tid < st) red[tid] = fmaxf(red[tid], red[tid + st]); __syncthreads(); }
    float mx = red[0]; __syncthreads();
    float sum = 0.0f;
    for (int s = tid; s < SRCLEN; s += 256) { float e = expf(sAtt[s] - mx); sAtt[s] = e; sum += e; }
    red[tid] = sum; __syncthreads();
    for (int st = 128; st > 0; st >>= 1) { if (tid < st) red[tid] += red[tid + st]; __syncthreads(); }
    float inv = 1.0f / red[0]; __syncthreads();

    for (int h = tid; h < HID; h += 256) {
        float c = 0.0f;
        for (int s = 0; s < SRCLEN; ++s) c += sAtt[s] * sf[(size_t)s * HID + h];
        c *= inv;
        decb[(size_t)blk * HID + h] = f32_to_bf16(tanhf(xr[h] + c));
    }
}

// logits = dec(80x768,bf16) @ W(768xVOCAB,bf16 packed) + b_gen, f32 out.
// One 16x16 tile per wave via v_wmma_f32_16x16x32_bf16, 8 waves/block.
__global__ void k_gemm(const unsigned short* __restrict__ decb,
                       const unsigned short* __restrict__ Wb,
                       const float* __restrict__ bgen,
                       float* __restrict__ logits) {
    int wave = threadIdx.x >> 5;
    int lane = threadIdx.x & 31;
    int nt = blockIdx.x * 8 + wave;
    int mt = blockIdx.y;
    if (nt >= NTILE) return;

    int half = lane >> 4;
    int arow = mt * 16 + (lane & 15);
    const unsigned short* ap = decb + (size_t)arow * HID + half * 8;
    const unsigned short* bp = Wb + (size_t)nt * KBLK * 512 + lane * 16;

    v8f acc = {};
    for (int kb = 0; kb < KBLK; ++kb) {
        union { v16bf v; v8bf h[2]; } A, B;
        A.h[0] = *(const v8bf*)(ap + kb * 32);
        A.h[1] = *(const v8bf*)(ap + kb * 32 + 16);
        B.v    = *(const v16bf*)(bp + (size_t)kb * 512);
        if (kb + 1 < KBLK)
            __builtin_prefetch(bp + (size_t)(kb + 1) * 512, 0, 1);
        acc = __builtin_amdgcn_wmma_f32_16x16x32_bf16(
            false, A.v, false, B.v, (short)0, acc, false, false);
    }

    int n = lane & 15;
    int col = nt * 16 + n;
    if (col < VOCAB) {
        float bb = bgen[col];
        int mbase = half * 8;
        for (int j = 0; j < 8; ++j) {
            int row = mt * 16 + mbase + j;
            logits[(size_t)row * VPAD + col] = acc[j] + bb;
        }
    }
}

// per-row max and log-sum-exp over the (unpadded) vocab
__global__ void k_rowstats(const float* __restrict__ logits, float* __restrict__ rstats) {
    int row = blockIdx.x, tid = threadIdx.x;
    __shared__ float red[256];
    float m = -INFINITY;
    for (int v = tid; v < VOCAB; v += 256) m = fmaxf(m, logits[(size_t)row * VPAD + v]);
    red[tid] = m; __syncthreads();
    for (int s = 128; s > 0; s >>= 1) { if (tid < s) red[tid] = fmaxf(red[tid], red[tid + s]); __syncthreads(); }
    float rmax = red[0]; __syncthreads();
    float sum = 0.0f;
    for (int v = tid; v < VOCAB; v += 256) sum += expf(logits[(size_t)row * VPAD + v] - rmax);
    red[tid] = sum; __syncthreads();
    for (int s = 128; s > 0; s >>= 1) { if (tid < s) red[tid] += red[tid + s]; __syncthreads(); }
    if (tid == 0) { rstats[row * 2] = rmax; rstats[row * 2 + 1] = logf(red[0]); }
}

// per-batch top-5 over K*V penalized candidate scores
__global__ void k_topk(const float* __restrict__ logits, const float* __restrict__ rstats,
                       const float* __restrict__ tlp, float* __restrict__ topkS,
                       int* __restrict__ topkI, int step) {
    int b = blockIdx.x, tid = threadIdx.x;
    float pen = powf((6.0f + (float)step) / 6.0f, ALPHA_P);
    float ls[5]; int li[5];
    for (int j = 0; j < 5; ++j) { ls[j] = -INFINITY; li[j] = 0; }
    for (int k = 0; k < NBEAM; ++k) {
        int row = b * NBEAM + k;
        float tk = tlp[row];
        float rmax = rstats[row * 2], rlse = rstats[row * 2 + 1];
        for (int v = tid; v < VOCAB; v += 256) {
            float lp = logits[(size_t)row * VPAD + v] - rmax - rlse;
            if (v == EOSTOK && step < MINLEN) lp = NEGBIG;
            float sc = (lp + tk) / pen;
            if (sc > ls[4]) {
                int id = k * VOCAB + v, j = 4;
                while (j > 0 && sc > ls[j - 1]) { ls[j] = ls[j - 1]; li[j] = li[j - 1]; --j; }
                ls[j] = sc; li[j] = id;
            }
        }
    }
    __shared__ float shS[256 * 5];
    __shared__ int   shI[256 * 5];
    for (int j = 0; j < 5; ++j) { shS[tid * 5 + j] = ls[j]; shI[tid * 5 + j] = li[j]; }
    __syncthreads();
    if (tid == 0) {
        float fs[5]; int fi[5];
        for (int j = 0; j < 5; ++j) { fs[j] = -INFINITY; fi[j] = 0; }
        for (int e = 0; e < 256 * 5; ++e) {
            float sc = shS[e];
            if (sc > fs[4]) {
                int id = shI[e], j = 4;
                while (j > 0 && sc > fs[j - 1]) { fs[j] = fs[j - 1]; fi[j] = fi[j - 1]; --j; }
                fs[j] = sc; fi[j] = id;
            }
        }
        for (int j = 0; j < 5; ++j) { topkS[b * 5 + j] = fs[j]; topkI[b * 5 + j] = fi[j]; }
    }
}

// beam bookkeeping (ping-pong alive buffers)
__global__ void k_update(const float* __restrict__ topkS, const int* __restrict__ topkI,
                         const int* __restrict__ aliveOld, int* __restrict__ aliveNew,
                         float* __restrict__ tlp, int* __restrict__ done,
                         float* __restrict__ bestS, int* __restrict__ bestP, int step) {
    int b = blockIdx.x, tid = threadIdx.x;
    __shared__ int stok[5], ssel[5], sfin[5];
    float pen = powf((6.0f + (float)step) / 6.0f, ALPHA_P);
    if (tid < 5) {
        int id = topkI[b * 5 + tid];
        int beam = id / VOCAB;
        int tok = id - beam * VOCAB;
        stok[tid] = tok;
        ssel[tid] = b * NBEAM + beam;
        sfin[tid] = (tok == EOSTOK) || (step == MAXLEN - 1);
        tlp[b * 5 + tid] = topkS[b * 5 + tid] * pen;
    }
    __syncthreads();
    for (int e = tid; e < NBEAM * (MAXLEN + 1); e += blockDim.x) {
        int k = e / (MAXLEN + 1), c = e % (MAXLEN + 1);
        int val = (c == step + 1) ? stok[k] : aliveOld[ssel[k] * (MAXLEN + 1) + c];
        aliveNew[(b * NBEAM + k) * (MAXLEN + 1) + c] = val;
    }
    __syncthreads();
    if (tid == 0) {
        int end_cond = sfin[0];
        int wasdone = done[b];
        float bc = -INFINITY; int bj = 0;
        for (int k = 0; k < 5; ++k) {
            int cand = (sfin[k] || end_cond) && !wasdone;
            float sc = cand ? topkS[b * 5 + k] : -INFINITY;
            if (sc > bc) { bc = sc; bj = k; }
        }
        if (bc > bestS[b]) {
            bestS[b] = bc;
            for (int t = 0; t < MAXLEN; ++t)
                bestP[b * MAXLEN + t] = aliveNew[(b * NBEAM + bj) * (MAXLEN + 1) + 1 + t];
        }
        done[b] = wasdone | end_cond;
    }
}

// outputs: best_pred (16*64 int32) then best_score (16 f32), concatenated flat
__global__ void k_final(const int* __restrict__ bestP, const float* __restrict__ bestS,
                        void* __restrict__ out) {
    int tid = blockIdx.x * 256 + threadIdx.x;
    int* ip = (int*)out;
    float* fp = (float*)out;
    if (tid < BATCH * MAXLEN) ip[tid] = bestP[tid];
    else if (tid < BATCH * MAXLEN + BATCH) fp[tid] = bestS[tid - BATCH * MAXLEN];
}

// ---------------- host side ----------------
extern "C" void kernel_launch(void* const* d_in, const int* in_sizes, int n_in,
                              void* d_out, int out_size, void* d_ws, size_t ws_size,
                              hipStream_t stream) {
    const int*   src   = (const int*)d_in[0];
    // d_in[1] = segs (unused by reference math)
    const int*   maskS = (const int*)d_in[2];
    const float* embed = (const float*)d_in[3];
    const float* Wgen  = (const float*)d_in[4];
    const float* bgen  = (const float*)d_in[5];

    char* base = (char*)d_ws;
    size_t off = 0;
    auto alloc = [&](size_t bytes) -> char* {
        char* p = base + off;
        off = (off + bytes + 255) & ~(size_t)255;
        return p;
    };
    float*          srcFeats = (float*)alloc((size_t)BATCH * SRCLEN * HID * 4);        // 25.2 MB
    unsigned short* Wb       = (unsigned short*)alloc((size_t)NTILE * KBLK * 512 * 2); // 46.9 MB
    float*          logits   = (float*)alloc((size_t)ROWS * VPAD * 4);                 //  9.8 MB
    float*          x        = (float*)alloc((size_t)ROWS * HID * 4);
    unsigned short* decb     = (unsigned short*)alloc((size_t)ROWS * HID * 2);
    float*          rstats   = (float*)alloc((size_t)ROWS * 2 * 4);
    float*          topkS    = (float*)alloc((size_t)BATCH * 5 * 4);
    int*            topkI    = (int*)alloc((size_t)BATCH * 5 * 4);
    int*            aliveA   = (int*)alloc((size_t)ROWS * (MAXLEN + 1) * 4);
    int*            aliveB   = (int*)alloc((size_t)ROWS * (MAXLEN + 1) * 4);
    float*          tlp      = (float*)alloc((size_t)ROWS * 4);
    int*            done     = (int*)alloc((size_t)BATCH * 4);
    float*          bestS    = (float*)alloc((size_t)BATCH * 4);
    int*            bestP    = (int*)alloc((size_t)BATCH * MAXLEN * 4);
    (void)ws_size; (void)n_in; (void)in_sizes; (void)out_size;

    {
        size_t total = (size_t)BATCH * SRCLEN * HID;
        k_src_feats<<<dim3((unsigned)((total + 255) / 256)), 256, 0, stream>>>(src, maskS, embed, srcFeats);
    }
    {
        size_t total = (size_t)NTILE * KBLK * 512;
        k_pack_w<<<dim3((unsigned)((total + 255) / 256)), 256, 0, stream>>>(Wgen, Wb);
    }
    k_init<<<dim3((ROWS * (MAXLEN + 1) + 255) / 256), 256, 0, stream>>>(aliveA, tlp, done, bestS, bestP);

    for (int step = 0; step < MAXLEN; ++step) {
        int* cur = (step & 1) ? aliveB : aliveA;
        int* nxt = (step & 1) ? aliveA : aliveB;
        k_gather_x<<<dim3(ROWS), 256, 0, stream>>>(cur, embed, x, step);
        k_attn<<<dim3(ROWS), 256, 0, stream>>>(x, srcFeats, maskS, decb);
        k_gemm<<<dim3((NTILE + 7) / 8, ROWS / 16), 256, 0, stream>>>(decb, Wb, bgen, logits);
        k_rowstats<<<dim3(ROWS), 256, 0, stream>>>(logits, rstats);
        k_topk<<<dim3(BATCH), 256, 0, stream>>>(logits, rstats, tlp, topkS, topkI, step);
        k_update<<<dim3(BATCH), 128, 0, stream>>>(topkS, topkI, cur, nxt, tlp, done, bestS, bestP, step);
    }
    k_final<<<dim3((BATCH * MAXLEN + BATCH + 255) / 256), 256, 0, stream>>>(bestP, bestS, d_out);
}
